// HGAKT_85753317032091
// MI455X (gfx1250) — matrix operand order repaired and checked by reading, not compiled, not run on hardware
//
#include <hip/hip_runtime.h>
#include <hip/hip_bf16.h>
#include <math.h>

// Problem constants (match reference)
#define BSZ 16
#define SEQ 512
#define DM  512
#define NH  8
#define DHD 64

typedef unsigned short u16;

// ---------------------------------------------------------------------------
// CDNA5 WMMA types
// ---------------------------------------------------------------------------
typedef __attribute__((ext_vector_type(16))) __bf16          v16bf;
typedef __attribute__((ext_vector_type(16))) unsigned short  v16u;
typedef __attribute__((ext_vector_type(8)))  unsigned short  v8u;
typedef __attribute__((ext_vector_type(8)))  float           v8f;

// fp32 -> bf16 round-to-nearest-even (bit trick); used only in elementwise
// producers and GEMM epilogues, never in the GEMM inner loop.
static __device__ __forceinline__ u16 f2bf(float f) {
    unsigned u = __float_as_uint(f);
    unsigned r = ((u >> 16) & 1u) + 0x7FFFu;
    return (u16)((u + r) >> 16);
}

// A-fragment, 16x32 bf16, A row-major [M,K] in bf16 (contiguous k = 16h+s, not ISA table order; B matches):
//   lanes 0-15 : row M=lane,    K = kb+0..15
//   lanes 16-31: row M=lane-16, K = kb+16..31
// Two contiguous 16B chunks -> two global_load_b128 per lane.
static __device__ __forceinline__ v16bf load_a_bf(const u16* __restrict__ A,
                                                  int lda, int mTile, int kb, int lane) {
    int m  = mTile + (lane & 15);
    int c0 = kb + ((lane >> 4) << 4);          // +0 or +16
    const u16* p = A + (size_t)m * lda + c0;
    v8u lo = *(const v8u*)p;
    v8u hi = *(const v8u*)(p + 8);
    v16u t = __builtin_shufflevector(lo, hi, 0,1,2,3,4,5,6,7,8,9,10,11,12,13,14,15);
    return __builtin_bit_cast(v16bf, t);
}

// B-fragment, 32x16 bf16 (contiguous k = 16h+s, same order as A):
//   lanes 0-15 : col N=lane,    rows K = kb+0..15
//   lanes 16-31: col N=lane-16, rows K = kb+16..31
// BT=true : B "weight style" [N,K] row-major -> one contiguous 32B read/lane
// BT=false: B [K,N] row-major -> 16 strided u16 reads/lane
template<bool BT>
static __device__ __forceinline__ v16bf load_b_bf(const u16* __restrict__ B,
                                                  int ldb, int nTile, int kb, int lane) {
    int n  = nTile + (lane & 15);
    int k0 = kb + ((lane >> 4) << 4);          // +0 or +16
    if (BT) {
        v16u t = *(const v16u*)(B + (size_t)n * ldb + k0);
        return __builtin_bit_cast(v16bf, t);
    } else {
        const u16* p = B + (size_t)k0 * ldb + n;
        v16u t;
#pragma unroll
        for (int j = 0; j < 16; ++j) t[j] = p[(size_t)j * ldb];
        return __builtin_bit_cast(v16bf, t);
    }
}

// ---------------------------------------------------------------------------
// Generic batched WMMA GEMM on bf16 operands, f32 accumulate:
//   C = scale * (A @ Bop) [+ bias] [+= C]
// One wave per 32x64 output tile: 2 A-fragments, every B fragment reused by
// two v_wmma_f32_16x16x32_bf16 (8 WMMA : 12 b128-loads per K-step).
// OUTBF selects f32 or bf16 output buffer. accFlag only valid with f32 out.
// ---------------------------------------------------------------------------
template<bool BT, bool OUTBF>
__global__ void __launch_bounds__(32)
wmma_gemm(const u16* __restrict__ A, const u16* __restrict__ B,
          const float* __restrict__ bias, void* __restrict__ Cv,
          int K, int lda, int ldb, int ldc,
          long long sAo, long long sAi, long long sBo, long long sBi,
          long long sCo, long long sCi, int bdiv,
          float scale, int accFlag)
{
    int z  = blockIdx.z;
    int zo = z / bdiv, zi = z - zo * bdiv;
    A += (size_t)zo * sAo + (size_t)zi * sAi;
    B += (size_t)zo * sBo + (size_t)zi * sBi;
    size_t cOff = (size_t)zo * sCo + (size_t)zi * sCi;

    int mTile = blockIdx.y << 5;               // 32 rows per wave
    int nTile = blockIdx.x << 6;               // 64 cols per wave
    int lane  = threadIdx.x & 31;

    v8f acc[2][4] = {};
    for (int kb = 0; kb < K; kb += 32) {
        v16bf a0 = load_a_bf(A, lda, mTile,      kb, lane);
        v16bf a1 = load_a_bf(A, lda, mTile + 16, kb, lane);
        if (kb + 32 < K) {  // stream next A K-slices (global_prefetch_b8)
            __builtin_prefetch(A + (size_t)(mTile +      (lane & 15)) * lda + kb + 32, 0, 1);
            __builtin_prefetch(A + (size_t)(mTile + 16 + (lane & 15)) * lda + kb + 32, 0, 1);
        }
#pragma unroll
        for (int t = 0; t < 4; ++t) {
            v16bf b = load_b_bf<BT>(B, ldb, nTile + (t << 4), kb, lane);
            acc[0][t] = __builtin_amdgcn_wmma_f32_16x16x32_bf16(false, a0, false, b, (short)0, acc[0][t], false, false);
            acc[1][t] = __builtin_amdgcn_wmma_f32_16x16x32_bf16(false, a1, false, b, (short)0, acc[1][t], false, false);
        }
    }

    // C/D layout: VGPR r -> row = base + r + 8*(lane>=16); col = nTile + (lane&15)
    int half = lane >> 4;
    int ncol = lane & 15;
#pragma unroll
    for (int g = 0; g < 2; ++g) {
#pragma unroll
        for (int t = 0; t < 4; ++t) {
            int col  = nTile + (t << 4) + ncol;
            float bv = bias ? bias[col] : 0.f;
#pragma unroll
            for (int r = 0; r < 8; ++r) {
                int row    = mTile + (g << 4) + r + (half << 3);
                size_t idx = cOff + (size_t)row * ldc + col;
                float val = acc[g][t][r] * scale + bv;
                if (OUTBF) {
                    ((u16*)Cv)[idx] = f2bf(val);
                } else {
                    float* C = (float*)Cv;
                    if (accFlag) val += C[idx];
                    C[idx] = val;
                }
            }
        }
    }
}

// ---------------------------------------------------------------------------
// fp32 -> bf16 buffer conversion (one pass per operand; float4 vectorized)
// ---------------------------------------------------------------------------
__global__ void cvt_f32_bf16(const float* __restrict__ s, u16* __restrict__ d, int n)
{
    int i = (blockIdx.x * blockDim.x + threadIdx.x) * 4;
    if (i + 3 < n) {
        float4 v = *(const float4*)(s + i);
        d[i]     = f2bf(v.x);
        d[i + 1] = f2bf(v.y);
        d[i + 2] = f2bf(v.z);
        d[i + 3] = f2bf(v.w);
    } else {
        for (; i < n; ++i) d[i] = f2bf(s[i]);
    }
}

// ---------------------------------------------------------------------------
// Distance-decay double-softmax + zero_pad row shift.
// One thread per OUTPUT row (b,h,i); source raw row r=i-1. Dual-writes f32 P
// (consumed by hterm_add) and bf16 Pbf (GEMM A operand). The 128 MB raw score
// tensor fits MI455X's 192 MB L2, so the multi-pass serial scan re-reads hit L2.
// ---------------------------------------------------------------------------
__global__ void decay_softmax_kernel(const float* __restrict__ raw,
                                     const float* __restrict__ gammas,
                                     float* __restrict__ P, u16* __restrict__ Pbf)
{
    int tid = blockIdx.x * blockDim.x + threadIdx.x;     // [0, BSZ*NH*SEQ)
    if (tid >= BSZ * NH * SEQ) return;
    int i  = tid % SEQ;
    int bh = tid / SEQ;
    int h  = bh % NH;
    size_t rowOff = (size_t)bh * SEQ * SEQ + (size_t)i * SEQ;
    float* dst  = P   + rowOff;
    u16*   dstb = Pbf + rowOff;
    int r = i - 1;
    if (i == 0 || r == 0) {
        for (int j = 0; j < SEQ; ++j) { dst[j] = 0.f; dstb[j] = 0; }
        return;
    }
    const float* x = raw + (size_t)bh * SEQ * SEQ + (size_t)r * SEQ;

    // masked softmax over j < r
    float m = -3.402823e38f;
    for (int j = 0; j < r; ++j) m = fmaxf(m, x[j]);
    float den = 0.f;
    for (int j = 0; j < r; ++j) den += __expf(x[j] - m);
    float inv = 1.f / den;

    float g     = gammas[h];
    float gamma = -log1pf(__expf(g));          // -softplus(gamma_h)

    // pass A: cumsum scan, track max of t = x*effect
    float cum = 0.f, mt = -3.402823e38f;
    for (int j = 0; j < r; ++j) {
        float p    = __expf(x[j] - m) * inv;
        cum += p;
        float rem  = 1.f - cum;                 // disttot == 1 for r >= 1
        float dist = sqrtf(fmaxf(rem * (float)(r - j), 0.f));
        float eff  = fminf(fmaxf(__expf(dist * gamma), 1e-5f), 1e5f);
        mt = fmaxf(mt, x[j] * eff);
    }
    // pass B: denominator of second softmax
    cum = 0.f;
    float den2 = 0.f;
    for (int j = 0; j < r; ++j) {
        float p    = __expf(x[j] - m) * inv;
        cum += p;
        float rem  = 1.f - cum;
        float dist = sqrtf(fmaxf(rem * (float)(r - j), 0.f));
        float eff  = fminf(fmaxf(__expf(dist * gamma), 1e-5f), 1e5f);
        den2 += __expf(x[j] * eff - mt);
    }
    float inv2 = 1.f / den2;
    // pass C: write both precisions
    cum = 0.f;
    for (int j = 0; j < r; ++j) {
        float p    = __expf(x[j] - m) * inv;
        cum += p;
        float rem  = 1.f - cum;
        float dist = sqrtf(fmaxf(rem * (float)(r - j), 0.f));
        float eff  = fminf(fmaxf(__expf(dist * gamma), 1e-5f), 1e5f);
        float val  = __expf(x[j] * eff - mt) * inv2;
        dst[j] = val; dstb[j] = f2bf(val);
    }
    for (int j = r; j < SEQ; ++j) { dst[j] = 0.f; dstb[j] = 0; }
}

// ---------------------------------------------------------------------------
// Strict-lower masked softmax over rows of f32 [batch,SEQ,SEQ] -> bf16 output
// (only consumer of the result is a WMMA GEMM A operand)
// ---------------------------------------------------------------------------
__global__ void masked_softmax_rows_bf(const float* __restrict__ sim,
                                       u16* __restrict__ out, int batch)
{
    int tid = blockIdx.x * blockDim.x + threadIdx.x;     // [0, batch*SEQ)
    if (tid >= batch * SEQ) return;
    int i = tid % SEQ;
    int b = tid / SEQ;
    const float* x = sim + (size_t)b * SEQ * SEQ + (size_t)i * SEQ;
    u16* o = out + (size_t)b * SEQ * SEQ + (size_t)i * SEQ;
    if (i == 0) {
        for (int j = 0; j < SEQ; ++j) o[j] = 0;
        return;
    }
    float m = -3.402823e38f;
    for (int j = 0; j < i; ++j) m = fmaxf(m, x[j]);
    float den = 0.f;
    for (int j = 0; j < i; ++j) den += __expf(x[j] - m);
    float inv = 1.f / den;
    for (int j = 0; j < i; ++j) o[j] = f2bf(__expf(x[j] - m) * inv);
    for (int j = i; j < SEQ; ++j) o[j] = 0;
}

// logd[i][j] = log(|i-j| + 1e-5) in f32 (for hterm) and bf16 (GEMM A operand)
__global__ void logd_fill(float* __restrict__ logd, u16* __restrict__ logdbf)
{
    int tid = blockIdx.x * blockDim.x + threadIdx.x;
    if (tid >= SEQ * SEQ) return;
    int i = tid / SEQ, j = tid % SEQ;
    float v = logf(fabsf((float)(i - j)) + 1e-5f);
    logd[tid] = v;
    logdbf[tid] = f2bf(v);
}

// Pbf[b,h,i,j] = bf16(P[b,h,i,j] + HT[b,i,j]*exp(-logd-atten))  (strict lower)
__global__ void hterm_add(const float* __restrict__ HT, const float* __restrict__ logd,
                          const float* __restrict__ atten,
                          const float* __restrict__ P, u16* __restrict__ Pbf)
{
    int j = blockIdx.x * blockDim.x + threadIdx.x;
    int i = blockIdx.y;
    int b = blockIdx.z;
    if (j >= i) return;                                   // strict lower triangle
    size_t ij = (size_t)i * SEQ + j;
    float t = HT[(size_t)b * SEQ * SEQ + ij] * __expf(-logd[ij] - atten[ij]);
    size_t base = (size_t)b * NH * SEQ * SEQ + ij;
#pragma unroll
    for (int h = 0; h < NH; ++h) {
        size_t idx = base + (size_t)h * SEQ * SEQ;
        Pbf[idx] = f2bf(P[idx] + t);
    }
}

// ---------------------------------------------------------------------------
// Launch. Workspace ~410 MiB (Hawkes-side buffers alias the 128 MiB raw-score
// region, which is dead after the decay kernel).
// ---------------------------------------------------------------------------
extern "C" void kernel_launch(void* const* d_in, const int* in_sizes, int n_in,
                              void* d_out, int out_size, void* d_ws, size_t ws_size,
                              hipStream_t stream)
{
    const float* q      = (const float*)d_in[0];
    const float* k      = (const float*)d_in[1];
    const float* v      = (const float*)d_in[2];
    const float* h_qkc  = (const float*)d_in[3];
    const float* h_qs   = (const float*)d_in[4];
    const float* Wk     = (const float*)d_in[5];
    const float* bk     = (const float*)d_in[6];
    const float* Wv     = (const float*)d_in[7];
    const float* bv     = (const float*)d_in[8];
    const float* Wo     = (const float*)d_in[9];
    const float* bo     = (const float*)d_in[10];
    const float* gammas = (const float*)d_in[11];
    const float* Wa     = (const float*)d_in[12];
    const float* ba     = (const float*)d_in[13];
    const float* Wa2    = (const float*)d_in[14];
    const float* ba2    = (const float*)d_in[15];
    const float* Wb     = (const float*)d_in[16];
    const float* bb     = (const float*)d_in[17];
    float* out = (float*)d_out;

    const size_t N_ACT = (size_t)BSZ * SEQ * DM;         // 4.19M elements
    const size_t N_W   = (size_t)DM * DM;                // 262K (== SEQ*SEQ)
    const size_t N_SC  = (size_t)BSZ * NH * SEQ * SEQ;   // 33.6M
    const size_t N_SIM = (size_t)BSZ * SEQ * SEQ;        // 4.19M

    // byte-offset carver, 256B aligned
    size_t off = 0;
    char* base = (char*)d_ws;
    auto carve = [&](size_t bytes) -> void* {
        void* p = base + off;
        off += (bytes + 255) & ~(size_t)255;
        return p;
    };
    // bf16 copies of inputs
    u16* qb   = (u16*)carve(N_ACT * 2);
    u16* kb2  = (u16*)carve(N_ACT * 2);
    u16* vb2  = (u16*)carve(N_ACT * 2);
    u16* hsb  = (u16*)carve(N_ACT * 2);
    u16* hkb  = (u16*)carve(N_ACT * 2);
    u16* Wkb  = (u16*)carve(N_W * 2);
    u16* Wvb  = (u16*)carve(N_W * 2);
    u16* Wob  = (u16*)carve(N_W * 2);
    u16* Wab  = (u16*)carve(N_W * 2);
    u16* Wa2b = (u16*)carve(N_W * 2);
    u16* Wbb  = (u16*)carve(N_W * 2);
    // bf16 projected activations
    u16* qhb = (u16*)carve(N_ACT * 2);
    u16* khb = (u16*)carve(N_ACT * 2);
    u16* vhb = (u16*)carve(N_ACT * 2);
    // score tensors
    float* raw = (float*)carve(N_SC * 4);   // raw qk scores (dead after decay)
    float* P   = (float*)carve(N_SC * 4);   // decayed+shifted scores, f32
    u16*   Pbf = (u16*)carve(N_SC * 2);     // bf16 twin for P@V GEMM
    // aliases into `raw` (only touched after decay_softmax_kernel):
    char* rb = (char*)raw;
    size_t ro = 0;
    auto carveR = [&](size_t bytes) -> void* {
        void* p = rb + ro;
        ro += (bytes + 255) & ~(size_t)255;
        return p;
    };
    float* simS   = (float*)carveR(N_SIM * 4);
    float* simK   = (float*)carveR(N_SIM * 4);
    u16*   hqsb   = (u16*)carveR(N_SIM * 2);
    u16*   hqkb   = (u16*)carveR(N_SIM * 2);
    float* HT     = (float*)carveR(N_SIM * 4);
    float* logd   = (float*)carveR((size_t)SEQ * SEQ * 4);
    u16*   logdbf = (u16*)carveR((size_t)SEQ * SEQ * 2);
    float* atten  = (float*)carveR((size_t)SEQ * SEQ * 4);
    u16*   conb   = (u16*)carveR(N_ACT * 2);

    dim3 wv(32);
    const long long ASD = (long long)SEQ * DM;
    const long long ASS = (long long)SEQ * SEQ;
    auto cvtGrid = [](size_t n) { return dim3((unsigned)((n / 4 + 255) / 256)); };

    // 0) one-time fp32 -> bf16 conversions of all GEMM operands
    cvt_f32_bf16<<<cvtGrid(N_ACT), 256, 0, stream>>>(q,     qb,   (int)N_ACT);
    cvt_f32_bf16<<<cvtGrid(N_ACT), 256, 0, stream>>>(k,     kb2,  (int)N_ACT);
    cvt_f32_bf16<<<cvtGrid(N_ACT), 256, 0, stream>>>(v,     vb2,  (int)N_ACT);
    cvt_f32_bf16<<<cvtGrid(N_ACT), 256, 0, stream>>>(h_qs,  hsb,  (int)N_ACT);
    cvt_f32_bf16<<<cvtGrid(N_ACT), 256, 0, stream>>>(h_qkc, hkb,  (int)N_ACT);
    cvt_f32_bf16<<<cvtGrid(N_W),   256, 0, stream>>>(Wk,  Wkb,  (int)N_W);
    cvt_f32_bf16<<<cvtGrid(N_W),   256, 0, stream>>>(Wv,  Wvb,  (int)N_W);
    cvt_f32_bf16<<<cvtGrid(N_W),   256, 0, stream>>>(Wo,  Wob,  (int)N_W);
    cvt_f32_bf16<<<cvtGrid(N_W),   256, 0, stream>>>(Wa,  Wab,  (int)N_W);
    cvt_f32_bf16<<<cvtGrid(N_W),   256, 0, stream>>>(Wa2, Wa2b, (int)N_W);
    cvt_f32_bf16<<<cvtGrid(N_W),   256, 0, stream>>>(Wb,  Wbb,  (int)N_W);

    // 1) projections -> bf16 activations (f32 accumulate + bias, bf16 store)
    dim3 gProj(DM / 64, (BSZ * SEQ) / 32, 1);
    wmma_gemm<true, true><<<gProj, wv, 0, stream>>>(qb,  Wkb, bk, qhb, DM, DM, DM, DM,
                                                    0,0,0,0,0,0, 1, 1.f, 0);
    wmma_gemm<true, true><<<gProj, wv, 0, stream>>>(kb2, Wkb, bk, khb, DM, DM, DM, DM,
                                                    0,0,0,0,0,0, 1, 1.f, 0);
    wmma_gemm<true, true><<<gProj, wv, 0, stream>>>(vb2, Wvb, bv, vhb, DM, DM, DM, DM,
                                                    0,0,0,0,0,0, 1, 1.f, 0);

    // 2) raw scores: per (b,h): qh_bh @ kh_bh^T / sqrt(DH)  -> f32
    dim3 gScore(SEQ / 64, SEQ / 32, BSZ * NH);
    wmma_gemm<true, false><<<gScore, wv, 0, stream>>>(qhb, khb, nullptr, raw, DHD, DM, DM, SEQ,
                                                      ASD, DHD, ASD, DHD,
                                                      (long long)NH * ASS, ASS, NH, 0.125f, 0);

    // 3) distance-decay double softmax + zero_pad shift -> P (f32) and Pbf (bf16)
    decay_softmax_kernel<<<(BSZ * NH * SEQ + 255) / 256, 256, 0, stream>>>(raw, gammas, P, Pbf);

    // 4) cosine-similarity Gram matrices (raw region now reusable)
    dim3 gCos(SEQ / 64, SEQ / 32, BSZ);
    wmma_gemm<true, false><<<gCos, wv, 0, stream>>>(hsb, hsb, nullptr, simS, DM, DM, DM, SEQ,
                                                    ASD, 0, ASD, 0, ASS, 0, 1, 1.f, 0);
    wmma_gemm<true, false><<<gCos, wv, 0, stream>>>(hkb, hkb, nullptr, simK, DM, DM, DM, SEQ,
                                                    ASD, 0, ASD, 0, ASS, 0, 1, 1.f, 0);
    masked_softmax_rows_bf<<<(BSZ * SEQ + 255) / 256, 256, 0, stream>>>(simS, hqsb, BSZ);
    masked_softmax_rows_bf<<<(BSZ * SEQ + 255) / 256, 256, 0, stream>>>(simK, hqkb, BSZ);

    // 5) HT = hqs@Wa^T + ba ; HT += hqk@Wa2^T + ba2   (f32)
    wmma_gemm<true, false><<<gCos, wv, 0, stream>>>(hqsb, Wab,  ba,  HT, SEQ, SEQ, SEQ, SEQ,
                                                    ASS, 0, 0, 0, ASS, 0, 1, 1.f, 0);
    wmma_gemm<true, false><<<gCos, wv, 0, stream>>>(hqkb, Wa2b, ba2, HT, SEQ, SEQ, SEQ, SEQ,
                                                    ASS, 0, 0, 0, ASS, 0, 1, 1.f, 1);

    // 6) atten = logd @ Wb^T + bb
    logd_fill<<<(SEQ * SEQ + 255) / 256, 256, 0, stream>>>(logd, logdbf);
    dim3 gAtt(SEQ / 64, SEQ / 32, 1);
    wmma_gemm<true, false><<<gAtt, wv, 0, stream>>>(logdbf, Wbb, bb, atten, SEQ, SEQ, SEQ, SEQ,
                                                    0,0,0,0,0,0, 1, 1.f, 0);

    // 7) Hawkes term folded into the bf16 score copy (broadcast across heads)
    hterm_add<<<dim3(SEQ / 256, SEQ, BSZ), 256, 0, stream>>>(HT, logd, atten, P, Pbf);

    // 8) attention output: concat[b,i,h*DH+d] = sum_j P[b,h,i,j]*vh[b,j,h*DH+d] -> bf16
    dim3 gOut(DHD / 64, SEQ / 32, BSZ * NH);
    wmma_gemm<false, true><<<gOut, wv, 0, stream>>>(Pbf, vhb, nullptr, conb, SEQ, SEQ, DM, DM,
                                                    (long long)NH * ASS, ASS, ASD, DHD,
                                                    ASD, DHD, NH, 1.f, 0);

    // 9) output projection: out = concat @ Wo^T + bo  (f32 result)
    wmma_gemm<true, false><<<gProj, wv, 0, stream>>>(conb, Wob, bo, out, DM, DM, DM, DM,
                                                     0,0,0,0,0,0, 1, 1.f, 0);
}
